// LSTM_39384850104606
// MI455X (gfx1250) — matrix operand order repaired and checked by reading, not compiled, not run on hardware
//
#include <hip/hip_runtime.h>

// LSTM: B=1024, T=512, IN=16, H=64, OUT=8, gates 4H=256, PyTorch gate order i,f,g,o
#define B_   1024
#define T_   512
#define IN_  16
#define H_   64
#define OUT_ 8

typedef _Float16 v16h __attribute__((ext_vector_type(16)));
typedef _Float16 v8h  __attribute__((ext_vector_type(8)));
typedef float    v8f  __attribute__((ext_vector_type(8)));

// CDNA5 has a hardware tanh (v_tanh_f32). Use it if the toolchain exposes it;
// otherwise fall back to v_exp_f32 + raw v_rcp_f32.
#if __has_builtin(__builtin_amdgcn_tanhf)
  #define TANHF(x) __builtin_amdgcn_tanhf(x)
#elif __has_builtin(__builtin_amdgcn_tanh_f32)
  #define TANHF(x) __builtin_amdgcn_tanh_f32(x)
#else
  #define TANHF(x) (1.0f - 2.0f * __builtin_amdgcn_rcpf(__expf(2.0f * (x)) + 1.0f))
#endif

// sigmoid(x) = 0.5*tanh(x/2) + 0.5  -> fma, tanh, fma
__device__ __forceinline__ float sigf(float x) {
    return __builtin_fmaf(0.5f, TANHF(0.5f * x), 0.5f);
}

__device__ __forceinline__ v16h cat8(v8h a, v8h b) {
    return __builtin_shufflevector(a, b, 0,1,2,3,4,5,6,7,8,9,10,11,12,13,14,15);
}

// One workgroup = 16 batch rows, 4 waves. Wave w owns h-columns [w*16, w*16+16).
// Per step, wave w computes gate tiles (16x16) for n = gg*64 + w*16 + (0..15), gg=0..3,
// so the LSTM cell update is wave-local. h is shared via double-buffered LDS (f16);
// one barrier per timestep.
__global__ __launch_bounds__(128) void lstm_fused_kernel(
    const float* __restrict__ x,    const float* __restrict__ W_ih,
    const float* __restrict__ W_hh, const float* __restrict__ b_ih,
    const float* __restrict__ b_hh, const float* __restrict__ W_fc,
    const float* __restrict__ b_fc, float* __restrict__ out)
{
    __shared__ float    xs[2][16][20];   // x tile, stride 80B (16B aligned, bank-spread)
    __shared__ _Float16 hs[2][16][80];   // h state f16, stride 160B (16B aligned, bank-spread)
    __shared__ float    hfin[16][H_];    // f32 h_last for the FC epilogue

    const int tid  = threadIdx.x;
    const int wave = tid >> 5;          // 0..3
    const int lane = tid & 31;
    const int lcol = lane & 15;         // N-column within 16-wide tile
    const int hi   = lane >> 4;         // half-wave selector
    const int b0   = blockIdx.x * 16;   // batch tile origin

    // ---- Preload f16 B operands (constant across the whole T loop) ----
    // B k order used here (contiguous, same as A): lanes 0-15 hold K=0..15, lanes 16-31 hold K=16..31,
    // element e of v16h = K = (hi?16:0)+e, column N = lane%16.
    v16h Bhh[4][2];   // h @ W_hh^T, K blocks of 32 (H=64 -> 2 blocks)
    v16h Bih[4];      // x @ W_ih^T, K=0..15 real, K=16..31 zero
    float biasg;      // g-gate bias (plain add before tanh)
    float bh[4];      // 0.5*bias for the sigmoid gates (folded into the 0.5x fma)
    #pragma unroll
    for (int gg = 0; gg < 4; ++gg) {
        const int n = gg * 64 + wave * 16 + lcol;   // gate column
        const float bsum = b_ih[n] + b_hh[n];
        bh[gg] = 0.5f * bsum;
        if (gg == 2) biasg = bsum;
        #pragma unroll
        for (int kb = 0; kb < 2; ++kb) {
            const float4* src = (const float4*)(W_hh + (size_t)n * H_ + kb * 32 + hi * 16);
            v16h b;
            #pragma unroll
            for (int q = 0; q < 4; ++q) {
                float4 v = src[q];
                b[4*q+0] = (_Float16)v.x; b[4*q+1] = (_Float16)v.y;
                b[4*q+2] = (_Float16)v.z; b[4*q+3] = (_Float16)v.w;
            }
            Bhh[gg][kb] = b;
        }
        v16h bi;
        {
            const float4* src = (const float4*)(W_ih + (size_t)n * IN_ + hi * 8);
            #pragma unroll
            for (int q = 0; q < 2; ++q) {
                float4 v = src[q];
                bi[4*q+0] = (_Float16)v.x; bi[4*q+1] = (_Float16)v.y;
                bi[4*q+2] = (_Float16)v.z; bi[4*q+3] = (_Float16)v.w;
            }
            #pragma unroll
            for (int e = 8; e < 16; ++e) bi[e] = (_Float16)0.f;
        }
        Bih[gg] = bi;
    }

    // x tile staging: 128 threads each load one float2 (row-major 16x16 tile)
    const int xrow = tid >> 3;
    const int xcol = (tid & 7) * 2;
    const float* xptr = x + ((size_t)(b0 + xrow)) * T_ * IN_ + xcol;

    // ---- Init: zero h buffer 0, zero c, stage x_0 into xs[0] ----
    for (int i = tid; i < 16 * 80; i += 128) ((_Float16*)hs[0])[i] = (_Float16)0.f;
    float c8[8];
    #pragma unroll
    for (int r = 0; r < 8; ++r) c8[r] = 0.f;
    {
        float2 xv = *(const float2*)(xptr);
        xs[0][xrow][xcol] = xv.x; xs[0][xrow][xcol + 1] = xv.y;
    }
    __syncthreads();

    const int arow = lcol;  // A-matrix row (M) held by this lane

    for (int t = 0; t < T_; ++t) {
        const int rb = t & 1;
        const int wb = rb ^ 1;

        // A operand from x: 16-bit A 16x32 layout; real K in [0,16), rest zero.
        v16h ax;
        {
            const float4* xp = (const float4*)&xs[rb][arow][hi * 8];
            float4 x0 = xp[0], x1 = xp[1];
            ax[0] = (_Float16)x0.x; ax[1] = (_Float16)x0.y;
            ax[2] = (_Float16)x0.z; ax[3] = (_Float16)x0.w;
            ax[4] = (_Float16)x1.x; ax[5] = (_Float16)x1.y;
            ax[6] = (_Float16)x1.z; ax[7] = (_Float16)x1.w;
            #pragma unroll
            for (int e = 8; e < 16; ++e) ax[e] = (_Float16)0.f;
        }
        // A operands from h (two K=32 blocks covering H=64)
        v16h ah0, ah1;
        {
            const v8h* hp = (const v8h*)&hs[rb][arow][hi * 16];  // 16B-aligned
            ah0 = cat8(hp[0], hp[1]);   // K kb0: 0..15 (16..31 for hi lanes)
            ah1 = cat8(hp[4], hp[5]);   // K kb1: 32..47 (48..63 for hi lanes)
        }

        // Stage x_{t+1} into the write buffer now; its global latency overlaps
        // a full step of WMMA + activation work. Safe: xs[wb] readers finished
        // before the previous end-of-iteration barrier.
        if (t + 1 < T_) {
            float2 xv = *(const float2*)(xptr + (size_t)(t + 1) * IN_);
            xs[wb][xrow][xcol] = xv.x; xs[wb][xrow][xcol + 1] = xv.y;
        }

        // Gates = x*W_ih^T + h*W_hh^T (f32 accumulation); 12 WMMAs, independent
        // accumulator chains across the 4 gates give the scheduler ILP to cover
        // WMMA->VALU hazards.
        v8f acc[4];
        #pragma unroll
        for (int gg = 0; gg < 4; ++gg) {
            v8f c = {};
            c = __builtin_amdgcn_wmma_f32_16x16x32_f16(false, ax,  false, Bih[gg],
                                                       (short)0, c, false, false);
            c = __builtin_amdgcn_wmma_f32_16x16x32_f16(false, ah0, false, Bhh[gg][0],
                                                       (short)0, c, false, false);
            c = __builtin_amdgcn_wmma_f32_16x16x32_f16(false, ah1, false, Bhh[gg][1],
                                                       (short)0, c, false, false);
            acc[gg] = c;
        }

        // Cell update, wave-local. C layout: VGPR r -> row M = hi*8 + r, col = lane%16.
        // Sigmoid gates: 0.5*tanh(0.5*acc + 0.5*bias) + 0.5 (bias folded into fma).
        #pragma unroll
        for (int r = 0; r < 8; ++r) {
            float ig = __builtin_fmaf(0.5f, TANHF(__builtin_fmaf(0.5f, acc[0][r], bh[0])), 0.5f);
            float fg = __builtin_fmaf(0.5f, TANHF(__builtin_fmaf(0.5f, acc[1][r], bh[1])), 0.5f);
            float gv = TANHF(acc[2][r] + biasg);
            float og = __builtin_fmaf(0.5f, TANHF(__builtin_fmaf(0.5f, acc[3][r], bh[3])), 0.5f);
            float cn = __builtin_fmaf(fg, c8[r], ig * gv);
            c8[r] = cn;
            float hn = og * TANHF(cn);
            hs[wb][hi * 8 + r][wave * 16 + lcol] = (_Float16)hn;
            if (t == T_ - 1) hfin[hi * 8 + r][wave * 16 + lcol] = hn;
        }

        __syncthreads();   // publish hs[wb] / xs[wb]; retire all reads of rb buffers
    }

    // ---- FC epilogue: out[b, 0, oc] = sigmoid(h_last @ W_fc^T + b_fc), 16x8 per WG ----
    {
        const int brow = tid >> 3;   // 0..15
        const int oc   = tid & 7;    // 0..7
        float a = b_fc[oc];
        const float* wf = W_fc + (size_t)oc * H_;
        #pragma unroll 8
        for (int k = 0; k < H_; ++k) a += hfin[brow][k] * wf[k];
        out[(size_t)(b0 + brow) * OUT_ + oc] = sigf(a);
    }
}

extern "C" void kernel_launch(void* const* d_in, const int* in_sizes, int n_in,
                              void* d_out, int out_size, void* d_ws, size_t ws_size,
                              hipStream_t stream) {
    const float* x    = (const float*)d_in[0];
    const float* W_ih = (const float*)d_in[1];
    const float* W_hh = (const float*)d_in[2];
    const float* b_ih = (const float*)d_in[3];
    const float* b_hh = (const float*)d_in[4];
    const float* W_fc = (const float*)d_in[5];
    const float* b_fc = (const float*)d_in[6];
    float* out = (float*)d_out;

    dim3 grid(B_ / 16);
    dim3 block(128);
    hipLaunchKernelGGL(lstm_fused_kernel, grid, block, 0, stream,
                       x, W_ih, W_hh, b_ih, b_hh, W_fc, b_fc, out);
}
